// SimpleCrossLingualAdapter_51908974739518
// MI455X (gfx1250) — compile-verified, matrix operand order checked by reading, not run on hardware
//
#include <hip/hip_runtime.h>
#include <hip/hip_bf16.h>

typedef __attribute__((ext_vector_type(16))) __bf16 v16bf;
typedef __attribute__((ext_vector_type(8)))  float  v8f;
typedef __attribute__((ext_vector_type(2)))  float  v2f;

// ---------------------------------------------------------------------------
// Problem dims: B=2048 T=128 C=32 H=128 G=384 E=256 BN=32 O=256 NL=3
// Fragment K-mapping for 16-bit WMMA operands (ISA 7.12.2):
//   half = lane>>4 ; e in [0,16) -> k = (e<8) ? 8*half+e : 8*half+e+8
// ---------------------------------------------------------------------------

__device__ __host__ __forceinline__ int kmap(int e, int half) {
    return (e < 8) ? (half * 8 + e) : (half * 8 + e + 8);
}

__device__ __forceinline__ float fast_sigmoid(float x) {
    float e = __expf(-x);          // v_exp_f32
#if __has_builtin(__builtin_amdgcn_rcpf)
    return __builtin_amdgcn_rcpf(1.f + e);   // v_rcp_f32
#else
    return 1.f / (1.f + e);
#endif
}

__device__ __forceinline__ float fast_tanh(float x) {
#if __has_builtin(__builtin_amdgcn_tanhf)
    return __builtin_amdgcn_tanhf(x);        // v_tanh_f32 (gfx1250 trans op)
#else
    return tanhf(x);
#endif
}

// Workspace layout (bytes)
#define OFF_WHH   0u                      // 2*24*4*512 bf16 = 196608
#define OFF_WIH   196608u                 // 2*24*512  bf16 = 49152
#define OFF_EMB   245760u                 // 2*128*128*512 bf16 = 33554432
#define OFF_PRJ   33800192u               // 16*64*32 float2 = 262144
#define OFF_FEAT  34062336u               // 2048*256 f32 = 2097152

// ---------------------------------------------------------------------------
// Prep: swizzle w_hh / w_ih into bf16 B-fragments, proj_w into f32 B-fragments.
// ---------------------------------------------------------------------------
__global__ void prep_weights_kernel(const float* __restrict__ w_ih_f,
                                    const float* __restrict__ w_hh_f,
                                    const float* __restrict__ w_ih_b,
                                    const float* __restrict__ w_hh_b,
                                    const float* __restrict__ proj_w,
                                    __bf16* __restrict__ whh_frag,
                                    __bf16* __restrict__ wih_frag,
                                    float2* __restrict__ prj_frag) {
    int idx = blockIdx.x * 256 + threadIdx.x;
    const int N_WHH = 2 * 24 * 4 * 512;   // 98304
    const int N_WIH = 2 * 24 * 512;       // 24576
    const int N_PRJ = 16 * 64 * 32;       // 32768 float2 slots
    if (idx < N_WHH) {
        int e = idx & 15, lane = (idx >> 4) & 31, kt = (idx >> 9) & 3;
        int rest = idx >> 11;
        int nt = rest % 24, dir = rest / 24;
        int half = lane >> 4, n = lane & 15;
        int g = nt * 16 + n;
        int k = kt * 32 + kmap(e, half);
        const float* w = dir ? w_hh_b : w_hh_f;   // [384,128]
        whh_frag[idx] = (__bf16)w[g * 128 + k];
    } else if (idx < N_WHH + N_WIH) {
        int i2 = idx - N_WHH;
        int e = i2 & 15, lane = (i2 >> 4) & 31;
        int rest = i2 >> 9;
        int nt = rest % 24, dir = rest / 24;
        int half = lane >> 4, n = lane & 15;
        int g = nt * 16 + n;
        int k = kmap(e, half);                    // K=32, single k-tile
        const float* w = dir ? w_ih_b : w_ih_f;   // [384,32]
        wih_frag[i2] = (__bf16)w[g * 32 + k];
    } else if (idx < N_WHH + N_WIH + N_PRJ) {
        int i3 = idx - N_WHH - N_WIH;
        int lane = i3 & 31, kt = (i3 >> 5) & 63, nt = i3 >> 11;
        int half = lane >> 4, n = lane & 15;
        int k = kt * 4 + 2 * half;
        int o = nt * 16 + n;
        prj_frag[i3] = make_float2(proj_w[o * 256 + k], proj_w[o * 256 + k + 1]);
    }
}

// ---------------------------------------------------------------------------
// Prep: token gather + embed, swizzled into bf16 A-fragment order.
// emb_frag layout: (((dir*128 + t)*128 + mt)*32 + lane)*16 + e
// ---------------------------------------------------------------------------
__global__ void prep_emb_kernel(const int* __restrict__ tokens,
                                const float* __restrict__ char_embed,
                                __bf16* __restrict__ emb_frag) {
    int idx = blockIdx.x * 256 + threadIdx.x;  // < 16777216
    int e    = idx & 15;
    int lane = (idx >> 4) & 31;
    int mt   = (idx >> 9) & 127;
    int t    = (idx >> 16) & 127;
    int dir  = idx >> 23;
    int half = lane >> 4;
    int b = mt * 16 + (lane & 15);
    int c = kmap(e, half);
    int tt = dir ? (127 - t) : t;
    int tok = tokens[b * 128 + tt];
    emb_frag[idx] = (__bf16)char_embed[tok * 32 + c];
}

// ---------------------------------------------------------------------------
// GRU kernel. grid = (B/32, 2 dirs), 256 threads (8 waves), 1 WG per WGP.
// Wave w owns gate columns j in [16w,16w+16): N-tiles {w, 8+w, 16+w}.
// Weight fragments register-resident (15 x v16bf = 120 VGPRs/wave).
// Hidden state f32 in registers; bf16 copy exchanged via ping-pong LDS
// fragment buffers -> single barrier per timestep.
// ---------------------------------------------------------------------------
__launch_bounds__(256, 1)
__global__ void gru_kernel(const __bf16* __restrict__ whh_frag,
                           const __bf16* __restrict__ wih_frag,
                           const __bf16* __restrict__ emb_frag,
                           const float* __restrict__ b_ih_f,
                           const float* __restrict__ b_hh_f,
                           const float* __restrict__ b_ih_b,
                           const float* __restrict__ b_hh_b,
                           float* __restrict__ feat) {
    __shared__ v16bf sHA[2][4 * 2 * 32];   // 16 KB: ping-pong h bf16 A-frags

    const int tid  = threadIdx.x;
    const int lane = tid & 31;
    const int wave = tid >> 5;
    const int dir  = blockIdx.y;
    const int mbase = blockIdx.x * 2;     // global batch m-tile base

    // ---- weight fragments -> registers (global loads, L2-hot) ----
    const v16bf* whhv = (const v16bf*)whh_frag + (size_t)dir * 24 * 4 * 32;
    const v16bf* wihv = (const v16bf*)wih_frag + (size_t)dir * 24 * 32;
    v16bf wR[4], wZ[4], wN[4];
    #pragma unroll
    for (int kt = 0; kt < 4; ++kt) {
        wR[kt] = whhv[(wave * 4 + kt) * 32 + lane];
        wZ[kt] = whhv[((8 + wave) * 4 + kt) * 32 + lane];
        wN[kt] = whhv[((16 + wave) * 4 + kt) * 32 + lane];
    }
    v16bf wRi = wihv[wave * 32 + lane];
    v16bf wZi = wihv[(8 + wave) * 32 + lane];
    v16bf wNi = wihv[(16 + wave) * 32 + lane];

    // ---- zero buffer 0 (read at t=0); every step fully rewrites its buffer
    {
        unsigned* z = (unsigned*)&sHA[0][0];
        for (int i = tid; i < 2048; i += 256) z[i] = 0u;
    }

    // ---- per-lane bias registers (gate column fixed per lane) ----
    const float* bih = dir ? b_ih_b : b_ih_f;
    const float* bhh = dir ? b_hh_b : b_hh_f;
    const int jcol = wave * 16 + (lane & 15);
    const float bR  = bih[jcol] + bhh[jcol];
    const float bZ  = bih[128 + jcol] + bhh[128 + jcol];
    const float bNi = bih[256 + jcol];
    const float bNh = bhh[256 + jcol];

    const int rowoff = (lane >> 4) * 8;
    // sHA write addressing: slot = (ktw*2+mt)*512 + half16*256 + m15*16 + ee
    const int kl     = (wave & 1) * 16 + (lane & 15);
    const int half16 = (kl >> 3) & 1;
    const int ee     = (kl & 7) + ((kl & 16) ? 8 : 0);
    const int wbase  = (wave >> 1) * 1024 + half16 * 256 + ee;
    __bf16* sHAh = (__bf16*)sHA;           // buffer stride = 4096 halfs

    v8f hreg[2];
    hreg[0] = (v8f){0.f, 0.f, 0.f, 0.f, 0.f, 0.f, 0.f, 0.f};
    hreg[1] = hreg[0];

    const v16bf* embv = (const v16bf*)(emb_frag + (size_t)dir * 128 * 128 * 512);
    v16bf eA[2], eAn[2];
    #pragma unroll
    for (int mt = 0; mt < 2; ++mt)
        eA[mt] = embv[(size_t)(mbase + mt) * 32 + lane];

    __syncthreads();

    #pragma clang loop unroll(disable)
    for (int t = 0; t < 128; ++t) {
        const int rb = t & 1;              // read buffer
        const int wb = rb ^ 1;             // write buffer

        // read previous-step h fragments (all k-tiles, both m-tiles)
        v16bf hA[4][2];
        #pragma unroll
        for (int kt = 0; kt < 4; ++kt)
            #pragma unroll
            for (int mt = 0; mt < 2; ++mt)
                hA[kt][mt] = sHA[rb][(kt * 2 + mt) * 32 + lane];

        // double-buffer next step's embedding fragments
        if (t + 1 < 128) {
            #pragma unroll
            for (int mt = 0; mt < 2; ++mt)
                eAn[mt] = embv[(size_t)((t + 1) * 128 + mbase + mt) * 32 + lane];
        }
        if (t + 2 < 128)
            __builtin_prefetch(&embv[(size_t)((t + 2) * 128 + mbase) * 32 + lane], 0, 1);

        #pragma unroll
        for (int mt = 0; mt < 2; ++mt) {
            v8f r  = (v8f){0.f,0.f,0.f,0.f,0.f,0.f,0.f,0.f};
            v8f z  = r, ng = r, nx = r;
            #pragma unroll
            for (int kt = 0; kt < 4; ++kt) {
                r  = __builtin_amdgcn_wmma_f32_16x16x32_bf16(false, hA[kt][mt], false,
                        wR[kt], (short)0, r,  false, false);
                z  = __builtin_amdgcn_wmma_f32_16x16x32_bf16(false, hA[kt][mt], false,
                        wZ[kt], (short)0, z,  false, false);
                ng = __builtin_amdgcn_wmma_f32_16x16x32_bf16(false, hA[kt][mt], false,
                        wN[kt], (short)0, ng, false, false);
            }
            r  = __builtin_amdgcn_wmma_f32_16x16x32_bf16(false, eA[mt], false,
                    wRi, (short)0, r,  false, false);
            z  = __builtin_amdgcn_wmma_f32_16x16x32_bf16(false, eA[mt], false,
                    wZi, (short)0, z,  false, false);
            nx = __builtin_amdgcn_wmma_f32_16x16x32_bf16(false, eA[mt], false,
                    wNi, (short)0, nx, false, false);

            // gate math in registers; publish new h bf16 fragment (write buf)
            #pragma unroll
            for (int v = 0; v < 8; ++v) {
                float rr = fast_sigmoid(r[v] + bR);
                float zz = fast_sigmoid(z[v] + bZ);
                float nn = fast_tanh(nx[v] + bNi + rr * (ng[v] + bNh));
                float h  = fmaf(zz, hreg[mt][v] - nn, nn);
                hreg[mt][v] = h;
                sHAh[wb * 4096 + wbase + mt * 512 + (rowoff + v) * 16] = (__bf16)h;
            }
        }
        eA[0] = eAn[0];
        eA[1] = eAn[1];
        __syncthreads();   // single barrier: publishes wb, retires rb
    }

    // ---- final hidden -> feat[B][E]: fwd cols 0..127, bwd 128..255 ----
    #pragma unroll
    for (int mt = 0; mt < 2; ++mt)
        #pragma unroll
        for (int v = 0; v < 8; ++v) {
            int m = mt * 16 + rowoff + v;
            feat[(size_t)(mbase * 16 + m) * 256 + dir * 128 + jcol] = hreg[mt][v];
        }
}

// ---------------------------------------------------------------------------
// Head kernel: language-routed adapter + LayerNorm, then final 256x256
// projection via full-precision f32 WMMA (16x16x4).  grid = B/32, 256 thr.
// ---------------------------------------------------------------------------
__launch_bounds__(256)
__global__ void head_kernel(const float* __restrict__ feat,
                            const int* __restrict__ lang_ids,
                            const float* __restrict__ down_w,
                            const float* __restrict__ down_b,
                            const float* __restrict__ up_w,
                            const float* __restrict__ up_b,
                            const float* __restrict__ ln_g,
                            const float* __restrict__ ln_b,
                            const float2* __restrict__ prj_frag,
                            const float* __restrict__ proj_b,
                            float* __restrict__ out) {
    __shared__ float sHid[32][32];
    __shared__ float sXN[32 * 260];   // padded stride vs 64 banks

    const int tid = threadIdx.x, lane = tid & 31, wave = tid >> 5;
    const int bbase = blockIdx.x * 32;

    // ---- phase A: bottleneck down-proj + exact GELU ----
    #pragma unroll 1
    for (int s = 0; s < 4; ++s) {
        int sl = wave * 4 + s;
        int b = bbase + sl;
        int lang = lang_ids[b];
        const float* fv = feat + (size_t)b * 256;
        const float* dw = down_w + ((size_t)lang * 32 + lane) * 256;
        float acc = down_b[lang * 32 + lane];
        #pragma unroll 8
        for (int e = 0; e < 256; e += 4) {
            float4 f4 = *(const float4*)(fv + e);
            float4 w4 = *(const float4*)(dw + e);
            acc += f4.x * w4.x + f4.y * w4.y + f4.z * w4.z + f4.w * w4.w;
        }
        sHid[sl][lane] = 0.5f * acc * (1.f + erff(acc * 0.70710678118654752f));
    }
    __syncthreads();

    // ---- phase B: up-proj + residual + LayerNorm (one wave per sample) ----
    #pragma unroll 1
    for (int s = 0; s < 4; ++s) {
        int sl = wave * 4 + s;
        int b = bbase + sl;
        int lang = lang_ids[b];
        float xv[8];
        float sum = 0.f, sumsq = 0.f;
        #pragma unroll
        for (int i = 0; i < 8; ++i) {
            int e = lane + 32 * i;
            float a = up_b[lang * 256 + e];
            const float* uw = up_w + ((size_t)(lang * 256 + e)) * 32;
            #pragma unroll
            for (int d = 0; d < 32; d += 4) {
                float4 w4 = *(const float4*)(uw + d);
                a += sHid[sl][d] * w4.x + sHid[sl][d + 1] * w4.y +
                     sHid[sl][d + 2] * w4.z + sHid[sl][d + 3] * w4.w;
            }
            float x = feat[(size_t)b * 256 + e] + a;
            xv[i] = x; sum += x; sumsq += x * x;
        }
        #pragma unroll
        for (int off = 16; off; off >>= 1) {
            sum   += __shfl_xor(sum, off, 32);
            sumsq += __shfl_xor(sumsq, off, 32);
        }
        float mu  = sum * (1.f / 256.f);
        float var = sumsq * (1.f / 256.f) - mu * mu;
        float inv = rsqrtf(var + 1e-5f);
        #pragma unroll
        for (int i = 0; i < 8; ++i) {
            int e = lane + 32 * i;
            sXN[sl * 260 + e] = (xv[i] - mu) * inv * ln_g[lang * 256 + e]
                                + ln_b[lang * 256 + e];
        }
    }
    __syncthreads();

    // ---- phase C: projection, f32 WMMA 16x16x4, K=256 (64 steps) ----
    const int half = lane >> 4, n = lane & 15, rowoff = half * 8;
    #pragma unroll 1
    for (int tile = wave; tile < 32; tile += 8) {
        int mt = tile & 1, nt = tile >> 1;
        v8f acc = (v8f){0.f,0.f,0.f,0.f,0.f,0.f,0.f,0.f};
        const float* arow = &sXN[(mt * 16 + n) * 260 + 2 * half];
        const float2* bptr = prj_frag + (size_t)nt * 64 * 32 + lane;
        #pragma unroll 8
        for (int kt = 0; kt < 64; ++kt) {
            v2f a;
            a.x = arow[kt * 4];
            a.y = arow[kt * 4 + 1];
            float2 bw = bptr[kt * 32];
            v2f bv; bv.x = bw.x; bv.y = bw.y;
            acc = __builtin_amdgcn_wmma_f32_16x16x4_f32(false, a, false, bv,
                                                        (short)0, acc, false, false);
        }
        int o = nt * 16 + n;
        float bias = proj_b[o];
        #pragma unroll
        for (int v = 0; v < 8; ++v) {
            int m = mt * 16 + rowoff + v;
            out[(size_t)(bbase + m) * 256 + o] = acc[v] + bias;
        }
    }
}

// ---------------------------------------------------------------------------
extern "C" void kernel_launch(void* const* d_in, const int* in_sizes, int n_in,
                              void* d_out, int out_size, void* d_ws, size_t ws_size,
                              hipStream_t stream) {
    const int*   tokens     = (const int*)  d_in[0];
    const int*   lang_ids   = (const int*)  d_in[1];
    const float* char_embed = (const float*)d_in[2];
    const float* w_ih_f     = (const float*)d_in[3];
    const float* w_hh_f     = (const float*)d_in[4];
    const float* b_ih_f     = (const float*)d_in[5];
    const float* b_hh_f     = (const float*)d_in[6];
    const float* w_ih_b     = (const float*)d_in[7];
    const float* w_hh_b     = (const float*)d_in[8];
    const float* b_ih_b     = (const float*)d_in[9];
    const float* b_hh_b     = (const float*)d_in[10];
    const float* down_w     = (const float*)d_in[11];
    const float* down_b     = (const float*)d_in[12];
    const float* up_w       = (const float*)d_in[13];
    const float* up_b       = (const float*)d_in[14];
    const float* ln_g       = (const float*)d_in[15];
    const float* ln_b       = (const float*)d_in[16];
    const float* proj_w     = (const float*)d_in[17];
    const float* proj_b     = (const float*)d_in[18];
    float* out = (float*)d_out;

    char* ws = (char*)d_ws;
    __bf16*  whh_frag = (__bf16*)(ws + OFF_WHH);
    __bf16*  wih_frag = (__bf16*)(ws + OFF_WIH);
    __bf16*  emb_frag = (__bf16*)(ws + OFF_EMB);
    float2*  prj_frag = (float2*)(ws + OFF_PRJ);
    float*   feat     = (float*) (ws + OFF_FEAT);

    (void)in_sizes; (void)n_in; (void)out_size; (void)ws_size;

    prep_weights_kernel<<<608, 256, 0, stream>>>(w_ih_f, w_hh_f, w_ih_b, w_hh_b,
                                                 proj_w, whh_frag, wih_frag, prj_frag);
    prep_emb_kernel<<<65536, 256, 0, stream>>>(tokens, char_embed, emb_frag);
    gru_kernel<<<dim3(64, 2), 256, 0, stream>>>(whh_frag, wih_frag, emb_frag,
                                                b_ih_f, b_hh_f, b_ih_b, b_hh_b, feat);
    head_kernel<<<64, 256, 0, stream>>>(feat, lang_ids, down_w, down_b,
                                        up_w, up_b, ln_g, ln_b,
                                        prj_frag, proj_b, out);
}
